// MultiheadAttention_39676907881438
// MI455X (gfx1250) — matrix operand order repaired and checked by reading, not compiled, not run on hardware
//
#include <hip/hip_runtime.h>

// ---------------------------------------------------------------------------
// FlashAttention forward, bf16 WMMA (gfx1250 / MI455X)
//   B=32, Nq=Nk=2048, D=128, fp32 in/out, fp32 softmax, bf16 matrix cores.
//   Round 3: K-dim permutation pi(j)=2*(j%16)+(j/16) on the P/V 32-key axis
//            -> packed b32 LDS stores for P and V staging (half the ds stores);
//            V-fragment preloads hoisted between S-WMMAs and softmax to fill
//            the WMMA->VALU hazard window and overlap LDS with softmax VALU.
// ---------------------------------------------------------------------------

typedef __attribute__((ext_vector_type(16))) __bf16 v16bf;
typedef __attribute__((ext_vector_type(8)))  float  v8f;

#define BATCH   32
#define SEQ     2048
#define HD      128
#define QTILE   128      // q rows per workgroup (8 waves x 16 rows)
#define JTILE   32       // keys per mainloop step
#define NJT     (SEQ / JTILE)
#define NWAVES  8
#define VSTR    40       // padded row stride (ushorts) of transposed V tile

struct U8 { unsigned u0,u1,u2,u3,u4,u5,u6,u7; };   // 32B POD for bit_cast -> v16bf

// fp32 -> bf16, round-to-nearest-even
static __device__ __forceinline__ unsigned short f2bf(float f) {
  unsigned u = __builtin_bit_cast(unsigned, f);
  u += 0x7FFFu + ((u >> 16) & 1u);
  return (unsigned short)(u >> 16);
}
static __device__ __forceinline__ unsigned pk(float a, float b) {
  return (unsigned)f2bf(a) | ((unsigned)f2bf(b) << 16);
}
// Build a 16-element bf16 fragment from two 16-byte chunks
static __device__ __forceinline__ v16bf ld2(const unsigned short* p0,
                                            const unsigned short* p1) {
  uint4 a = *(const uint4*)p0;
  uint4 b = *(const uint4*)p1;
  U8 t{a.x, a.y, a.z, a.w, b.x, b.y, b.z, b.w};
  return __builtin_bit_cast(v16bf, t);
}

// XOR butterfly within 16-lane rows via v_permlane16_b32 (VALU, no LDS).
static __device__ __forceinline__ float pl16(float v, unsigned lo, unsigned hi) {
  unsigned u = __builtin_bit_cast(unsigned, v);
  unsigned p = __builtin_amdgcn_permlane16(u, u, lo, hi, false, false);
  return __builtin_bit_cast(float, p);
}
static __device__ __forceinline__ float rmax16(float t) {
  t = fmaxf(t, pl16(t, 0x67452301u, 0xEFCDAB89u));   // xor 1
  t = fmaxf(t, pl16(t, 0x54761032u, 0xDCFE98BAu));   // xor 2
  t = fmaxf(t, pl16(t, 0x32107654u, 0xBA98FEDCu));   // xor 4
  t = fmaxf(t, pl16(t, 0xFEDCBA98u, 0x76543210u));   // xor 8
  return t;
}
static __device__ __forceinline__ float rsum16(float t) {
  t += pl16(t, 0x67452301u, 0xEFCDAB89u);
  t += pl16(t, 0x54761032u, 0xDCFE98BAu);
  t += pl16(t, 0x32107654u, 0xBA98FEDCu);
  t += pl16(t, 0xFEDCBA98u, 0x76543210u);
  return t;
}

extern "C" __global__ void __launch_bounds__(256, 1)
fa_fwd_bf16wmma(const float* __restrict__ Qg, const float* __restrict__ Kg,
                const float* __restrict__ Vg, float* __restrict__ Og) {
  __shared__ unsigned short sK [2][JTILE * HD];      // 2 x 8KB, row-major bf16
  __shared__ unsigned short sVt[2][HD * VSTR];       // 2 x 10KB, sVt[d][pi(j)]
  __shared__ unsigned short sP [NWAVES][16 * JTILE]; // 8KB, cols in pi(j) order

  const int tid   = threadIdx.x;
  const int batch = blockIdx.y;
  const int qt    = blockIdx.x;
  const int wv    = tid >> 5;
  const int lane  = tid & 31;
  const int hs    = lane >> 4;   // half-wave select
  const int l15   = lane & 15;

  // K staging coords: 32 rows x (8 threads x 16 floats)
  const int srow = tid >> 3;
  const int scol = (tid & 7) * 16;
  // V staging coords: thread handles rows (vjn, vjn+16), d-range [vd, vd+8)
  const int vjn = tid & 15;
  const int vd  = (tid >> 4) * 8;
  const size_t kvbase = (size_t)batch * SEQ * HD;

  const float qscale = 1.4426950408889634f / 11.313708498984761f; // log2e/sqrt(128)
  const int qrow0 = qt * QTILE + wv * 16;

  // ---- Load Q once into WMMA A-fragments (pre-scaled, bf16) ----
  v16bf aq[4];
  {
    const float* qp = Qg + ((size_t)batch * SEQ + qrow0 + l15) * HD + hs * 16;
#pragma unroll
    for (int kc = 0; kc < 4; ++kc) {
      const float4* q4 = (const float4*)(qp + kc * 32);
      float4 f0 = q4[0], f1 = q4[1];   // K = base+0..7
      float4 f2 = q4[2], f3 = q4[3];   // K = base+8..15
      U8 t{pk(f0.x*qscale, f0.y*qscale), pk(f0.z*qscale, f0.w*qscale),
           pk(f1.x*qscale, f1.y*qscale), pk(f1.z*qscale, f1.w*qscale),
           pk(f2.x*qscale, f2.y*qscale), pk(f2.z*qscale, f2.w*qscale),
           pk(f3.x*qscale, f3.y*qscale), pk(f3.z*qscale, f3.w*qscale)};
      aq[kc] = __builtin_bit_cast(v16bf, t);
    }
  }

  // ---- Accumulators ----
  v8f oacc[8];
  float m[8], l[8];
#pragma unroll
  for (int b = 0; b < 8; ++b)
#pragma unroll
    for (int r = 0; r < 8; ++r) oacc[b][r] = 0.0f;
#pragma unroll
  for (int r = 0; r < 8; ++r) { m[r] = -__builtin_inff(); l[r] = 0.0f; }

  // ---- Stage one 32-key K/V tile: global fp32 -> bf16 -> LDS ----
  // K row-major; V transposed with permuted key axis: sVt[d][2*(j%16)+(j/16)]
  auto stage = [&](int jt, int buf) {
    const float* kp = Kg + kvbase + (size_t)(jt * JTILE + srow) * HD + scol;
    float4 k0 = ((const float4*)kp)[0], k1 = ((const float4*)kp)[1];
    float4 k2 = ((const float4*)kp)[2], k3 = ((const float4*)kp)[3];
    uint4* dk = (uint4*)&sK[buf][srow * HD + scol];
    dk[0] = make_uint4(pk(k0.x,k0.y), pk(k0.z,k0.w), pk(k1.x,k1.y), pk(k1.z,k1.w));
    dk[1] = make_uint4(pk(k2.x,k2.y), pk(k2.z,k2.w), pk(k3.x,k3.y), pk(k3.z,k3.w));

    const float* vpa = Vg + kvbase + (size_t)(jt * JTILE + vjn) * HD + vd;
    const float* vpb = vpa + 16 * HD;
    float4 a0 = ((const float4*)vpa)[0], a1 = ((const float4*)vpa)[1];
    float4 b0 = ((const float4*)vpb)[0], b1 = ((const float4*)vpb)[1];
    float va[8] = {a0.x,a0.y,a0.z,a0.w, a1.x,a1.y,a1.z,a1.w};
    float vb[8] = {b0.x,b0.y,b0.z,b0.w, b1.x,b1.y,b1.z,b1.w};
#pragma unroll
    for (int i = 0; i < 8; ++i)   // packed pair (j, j+16) -> one b32 store
      *(unsigned*)&sVt[buf][(vd + i) * VSTR + 2 * vjn] = pk(va[i], vb[i]);
  };

  // ---- One 32-key flash-attention step ----
  auto compute = [&](int buf) {
    // Preload all 8 K B-fragments (16 x ds_load_b128 in flight)
    v16bf bk[8];
#pragma unroll
    for (int kc = 0; kc < 4; ++kc) {
      const unsigned short* kb0 = &sK[buf][ l15       * HD + kc * 32 + hs * 16];
      const unsigned short* kb1 = &sK[buf][(16 + l15) * HD + kc * 32 + hs * 16];
      bk[kc * 2]     = ld2(kb0, kb0 + 8);
      bk[kc * 2 + 1] = ld2(kb1, kb1 + 8);
    }
    v8f c0, c1;                                    // S tile 16x32
#pragma unroll
    for (int r = 0; r < 8; ++r) { c0[r] = 0.0f; c1[r] = 0.0f; }
#pragma unroll
    for (int kc = 0; kc < 4; ++kc) {
      c0 = __builtin_amdgcn_wmma_f32_16x16x32_bf16(false, aq[kc], false, bk[kc*2],
                                                   (short)0, c0, false, false);
      c1 = __builtin_amdgcn_wmma_f32_16x16x32_bf16(false, aq[kc], false, bk[kc*2+1],
                                                   (short)0, c1, false, false);
    }

    // Preload all 8 V B-fragments NOW: independent of c0/c1, fills the
    // WMMA->VALU hazard window and overlaps LDS pipe with softmax VALU.
    v16bf bv[8];
#pragma unroll
    for (int b = 0; b < 8; ++b) {
      const unsigned short* vfp = &sVt[buf][(b * 16 + l15) * VSTR + hs * 16];
      bv[b] = ld2(vfp, vfp + 8);
    }

    // online softmax (base-2 domain), VALU-only cross-lane reductions
    float alpha[8];
#pragma unroll
    for (int r = 0; r < 8; ++r) {
      const float t  = rmax16(fmaxf(c0[r], c1[r]));
      const float mn = fmaxf(m[r], t);
      alpha[r] = exp2f(m[r] - mn);
      c0[r] = exp2f(c0[r] - mn);
      c1[r] = exp2f(c1[r] - mn);
      const float s = rsum16(c0[r] + c1[r]);
      l[r] = l[r] * alpha[r] + s;
      m[r] = mn;
    }
#pragma unroll
    for (int b = 0; b < 8; ++b)
#pragma unroll
      for (int r = 0; r < 8; ++r) oacc[b][r] *= alpha[r];

    // P -> wave-private LDS in pi(j) column order: (c0,c1) pair is adjacent,
    // one packed b32 store per row instead of two b16 stores.
    unsigned* pw32 = (unsigned*)&sP[wv][0];
#pragma unroll
    for (int r = 0; r < 8; ++r) {
      const int row = hs * 8 + r;
      pw32[row * 16 + l15] = pk(c0[r], c1[r]);
    }
    const unsigned short* pr = &sP[wv][l15 * JTILE + hs * 16];
    v16bf pa = ld2(pr, pr + 8);

    // O += P (16x32) @ V (32x128), K-dim in pi(j) order on both operands
#pragma unroll
    for (int b = 0; b < 8; ++b)
      oacc[b] = __builtin_amdgcn_wmma_f32_16x16x32_bf16(false, pa, false, bv[b],
                                                        (short)0, oacc[b], false, false);
  };

  // ---- Main loop: double-buffered, 1 barrier per 32-key step ----
  stage(0, 0);
  __syncthreads();
  for (int jt = 0; jt < NJT; ++jt) {
    const int cur = jt & 1;
    if (jt + 1 < NJT) stage(jt + 1, cur ^ 1);
    if (jt + 2 < NJT) {   // L2 prefetch for tile after next (global_prefetch_b8)
      __builtin_prefetch(Kg + kvbase + (size_t)((jt + 2) * JTILE + srow) * HD + scol, 0, 1);
      __builtin_prefetch(Vg + kvbase + (size_t)((jt + 2) * JTILE + vjn)  * HD + vd,   0, 1);
    }
    compute(cur);
    __syncthreads();
  }

  // ---- Epilogue: O /= l, write fp32 (coalesced per 16-lane half) ----
#pragma unroll
  for (int r = 0; r < 8; ++r) {
    const float inv = 1.0f / l[r];
    float* orow = Og + ((size_t)batch * SEQ + qrow0 + hs * 8 + r) * HD + l15;
#pragma unroll
    for (int b = 0; b < 8; ++b) orow[b * 16] = oacc[b][r] * inv;
  }
}

extern "C" void kernel_launch(void* const* d_in, const int* in_sizes, int n_in,
                              void* d_out, int out_size, void* d_ws, size_t ws_size,
                              hipStream_t stream) {
  const float* Q = (const float*)d_in[0];
  const float* K = (const float*)d_in[1];
  const float* V = (const float*)d_in[2];
  float* O = (float*)d_out;
  dim3 grid(SEQ / QTILE, BATCH);   // 16 q-tiles x 32 batches = 512 workgroups
  fa_fwd_bf16wmma<<<grid, dim3(256), 0, stream>>>(Q, K, V, O);
}
